// Swarm6502_21208548507705
// MI455X (gfx1250) — matrix operand order repaired and checked by reading, not compiled, not run on hardware
//
#include <hip/hip_runtime.h>

#define NS 65536
#define HID 512
#define NOUT 1296

typedef _Float16 v16h __attribute__((ext_vector_type(16)));
typedef _Float16 v8h  __attribute__((ext_vector_type(8)));
typedef float    v8f  __attribute__((ext_vector_type(8)));

struct IntPtrs  { const int*   p[9]; };
struct BiasPtrs { const float* p[5]; };

// expert metadata
__constant__ int c_idx_src[5][6] = {
  {0,8,255,7,0,0},   // alu:   A, Val, carry, Op
  {0,8,255,7,0,0},   // logic: A, Val, carry, Op
  {0,1,2,8,7,0},     // move:  A, X, Y, Val, Op
  {5,6,4,8,3,7},     // flow:  PCH, PCL, P, Val, SP, Op
  {0,1,3,4,8,7}      // stack: A, X, SP, P, Val, Op
};
__constant__ int c_k[5]      = {4,4,5,6,6};
__constant__ int c_kD[5]     = {256,256,320,384,384};
__constant__ int c_embOff[5] = {0,65536,131072,212992,311296};   // halves
__constant__ int c_w1Off[5]  = {0,131072,262144,425984,622592};  // halves
__constant__ int c_w2Off[5]  = {0,139264,278528,409600,540672};  // halves
__constant__ int c_w2P[5]    = {272,272,256,256,256};            // padded out cols
__constant__ int c_width[5]  = {264,264,256,256,256};
__constant__ int c_oOff[5]   = {0,264,528,784,1040};

__global__ void k_zero(int* cnt){ if (threadIdx.x < 5) cnt[threadIdx.x] = 0; }

__global__ void k_cvt(const float* __restrict__ s, _Float16* __restrict__ d, int n){
  int i = blockIdx.x*blockDim.x + threadIdx.x;
  if (i < n) d[i] = (_Float16)s[i];
}

// W1 (kD x 512) f32 -> W1t (512 x kD) f16
__global__ void k_w1t(const float* __restrict__ s, _Float16* __restrict__ d, int kD){
  int i = blockIdx.x*blockDim.x + threadIdx.x;
  int n = HID*kD;
  if (i < n){ int r = i / kD, k = i - r*kD; d[i] = (_Float16)s[k*HID + r]; }
}

// Wr (512 x 256) [+ Wf (512 x 8)] f32 -> packed transposed (P x 512) f16, zero pad
__global__ void k_w2t(const float* __restrict__ wr, const float* __restrict__ wf,
                      _Float16* __restrict__ d, int P){
  int i = blockIdx.x*blockDim.x + threadIdx.x;
  int n = P*HID;
  if (i < n){
    int r = i / HID, h = i - r*HID;
    float v = 0.f;
    if (r < 256) v = wr[h*256 + r];
    else if (wf != nullptr && r < 264) v = wf[h*8 + (r-256)];
    d[i] = (_Float16)v;
  }
}

__global__ void k_compact(const int* __restrict__ Op, const int* __restrict__ fu,
                          int* __restrict__ cnt, int* __restrict__ idx){
  int i = blockIdx.x*blockDim.x + threadIdx.x;
  if (i < NS){
    int e = fu[Op[i] & 255];
    if (e >= 0 && e < 5){
      int s = atomicAdd(&cnt[e], 1);
      idx[e*NS + s] = i;
    }
  }
}

static __device__ __forceinline__ v8f wmma32f16(v16h a, v16h b, v8f c){
  return __builtin_amdgcn_wmma_f32_16x16x32_f16(false, a, false, b, (short)0, c, false, false);
}

// A-fragment (16-bit, 16x32): per lane K = {koff..koff+7, 16+koff..23+koff}
static __device__ __forceinline__ v16h ld_afrag(const _Float16* p){
  v8h lo = *(const v8h*)p;
  v8h hi = *(const v8h*)(p + 16);
  return __builtin_shufflevector(lo, hi, 0,1,2,3,4,5,6,7,8,9,10,11,12,13,14,15);
}

static __device__ __forceinline__ float gelu_t(float x){
  float x3 = x*x*x;
  return 0.5f*x*(1.0f + tanhf(0.7978845608f*(x + 0.044715f*x3)));
}

__launch_bounds__(256)
__global__ void k_moe(IntPtrs ip, BiasPtrs bpv, const int* __restrict__ cnt,
                      const int* __restrict__ idxbuf,
                      const _Float16* __restrict__ emb,
                      const _Float16* __restrict__ w1t,
                      const _Float16* __restrict__ w2t,
                      float* __restrict__ out){
  constexpr int AS = 392;   // A LDS row stride (halves), 16B-aligned rows
  constexpr int HS = 520;   // hidden LDS row stride (halves)
  extern __shared__ char smem[];
  _Float16* Al  = (_Float16*)smem;                 // 64*392*2 = 50176 B
  _Float16* Hl  = (_Float16*)(smem + 50176);       // 64*520*2 = 66560 B
  int*      sid = (int*)(smem + 116736);           // 256 B
  int*      ids = (int*)(smem + 116992);           // 64*6*4 = 1536 B (end 118528)

  const int e    = blockIdx.y;
  const int cntE = cnt[e];
  const int base = blockIdx.x * 64;
  if (base >= cntE) return;
  const int nv = min(64, cntE - base);

  const int tid  = threadIdx.x;
  const int lane = tid & 31;
  const int wv   = tid >> 5;
  const int kD   = c_kD[e];
  const int ke   = c_k[e];
  const _Float16* embE = emb + c_embOff[e];
  const _Float16* w1E  = w1t + c_w1Off[e];
  const _Float16* w2E  = w2t + c_w2Off[e];
  const float*    b1   = bpv.p[e];
  const int P2    = c_w2P[e];
  const int width = c_width[e];
  const int oOff  = c_oOff[e];

  if (tid < 64) sid[tid] = idxbuf[e*NS + base + (tid < nv ? tid : 0)];
  __syncthreads();

  // per-row embedding indices
  for (int g = tid; g < 64*6; g += 256){
    int r = g / 6, j = g - r*6;
    if (j < ke){
      int src = c_idx_src[e][j];
      int s = sid[r];
      ids[r*6 + j] = (src == 255) ? (ip.p[4][s] & 1) : (ip.p[src][s] & 255);
    }
  }
  __syncthreads();

  // gather features (f16) into Al: row r, col c = j*64 + d
  const int gpr = kD >> 2;  // 8-byte groups per row
  for (int g = tid; g < 64*gpr; g += 256){
    int r = g / gpr, c = (g - r*gpr) << 2;
    int j = c >> 6, d = c & 63;
    const _Float16* sp = embE + (j << 14) + (ids[r*6 + j] << 6) + d;
    *(uint2*)(Al + r*AS + c) = *(const uint2*)sp;
  }
  __syncthreads();

  const int row16 = lane & 15;
  const int kofA  = (lane < 16) ? 0 : 8;
  const int kofB  = (lane < 16) ? 0 : 8;
  const int hiAdd = (lane < 16) ? 0 : 8;
  const int nK1   = kD >> 5;

  // ---- GEMM1: (64 x kD) @ (kD x 512) + bias, GELU -> Hl (f16) ----
  for (int nt = 0; nt < 4; ++nt){
    const int nb = (wv*4 + nt) << 4;
    v8f acc[4] = {};
    const _Float16* bpB = w1E + (nb + row16)*kD + kofB;
    for (int kk = 0; kk < nK1; ++kk){
      const int kb = kk << 5;
      v16h bf = ld_afrag(bpB + kb);
      #pragma unroll
      for (int m = 0; m < 4; ++m){
        v16h af = ld_afrag(Al + (m*16 + row16)*AS + kb + kofA);
        acc[m] = wmma32f16(af, bf, acc[m]);
      }
    }
    const float bias = b1[nb + row16];
    #pragma unroll
    for (int m = 0; m < 4; ++m){
      #pragma unroll
      for (int v = 0; v < 8; ++v){
        float x = acc[m][v] + bias;
        Hl[(m*16 + v + hiAdd)*HS + nb + row16] = (_Float16)gelu_t(x);
      }
    }
  }
  __syncthreads();

  // ---- GEMM2: (64 x 512) @ (512 x P2) -> scatter to out ----
  const int nT2 = P2 >> 4;           // 16 or 17 tiles
  for (int ti = wv; ti < nT2; ti += 8){
    const int nb = ti << 4;
    v8f acc[4] = {};
    const _Float16* bpB = w2E + (nb + row16)*HID + kofB;
    for (int kk = 0; kk < 16; ++kk){
      const int kb = kk << 5;
      v16h bf = ld_afrag(bpB + kb);
      #pragma unroll
      for (int m = 0; m < 4; ++m){
        v16h af = ld_afrag(Hl + (m*16 + row16)*HS + kb + kofA);
        acc[m] = wmma32f16(af, bf, acc[m]);
      }
    }
    const int col = nb + row16;
    if (col < width){
      #pragma unroll
      for (int m = 0; m < 4; ++m){
        #pragma unroll
        for (int v = 0; v < 8; ++v){
          int r = m*16 + v + hiAdd;
          if (r < nv) out[(size_t)sid[r]*NOUT + oOff + col] = acc[m][v];
        }
      }
    }
  }

  // zero-fill inactive expert columns (16B vector stores; offsets are /8-aligned)
  for (int g = tid; g < 64*324; g += 256){
    int r = g / 324, c = (g - r*324) << 2;
    if (r < nv && (c < oOff || c >= oOff + width)){
      float4 z = make_float4(0.f, 0.f, 0.f, 0.f);
      *(float4*)(out + (size_t)sid[r]*NOUT + c) = z;
    }
  }
}

extern "C" void kernel_launch(void* const* d_in, const int* in_sizes, int n_in,
                              void* d_out, int out_size, void* d_ws, size_t ws_size,
                              hipStream_t stream) {
  (void)in_sizes; (void)n_in; (void)out_size; (void)ws_size;

  const int* Op = (const int*)d_in[7];
  const int* fu = (const int*)d_in[9];

  const float* emb_src[5] = {(const float*)d_in[10], (const float*)d_in[13],
                             (const float*)d_in[16], (const float*)d_in[19],
                             (const float*)d_in[22]};
  const float* w1_src[5]  = {(const float*)d_in[11], (const float*)d_in[14],
                             (const float*)d_in[17], (const float*)d_in[20],
                             (const float*)d_in[23]};
  const float* b1_src[5]  = {(const float*)d_in[12], (const float*)d_in[15],
                             (const float*)d_in[18], (const float*)d_in[21],
                             (const float*)d_in[24]};
  const float* wr_src[5]  = {(const float*)d_in[25], (const float*)d_in[27],
                             (const float*)d_in[29], (const float*)d_in[30],
                             (const float*)d_in[31]};
  const float* wf_src[5]  = {(const float*)d_in[26], (const float*)d_in[28],
                             nullptr, nullptr, nullptr};

  static const int kH[5]      = {4,4,5,6,6};
  static const int kDH[5]     = {256,256,320,384,384};
  static const int P2H[5]     = {272,272,256,256,256};
  static const int embOffH[5] = {0,65536,131072,212992,311296};
  static const int w1OffH[5]  = {0,131072,262144,425984,622592};
  static const int w2OffH[5]  = {0,139264,278528,409600,540672};

  char* w = (char*)d_ws;
  int*      cnt   = (int*)w;
  int*      idx   = (int*)(w + 64);
  _Float16* emb16 = (_Float16*)(w + 1310784);
  _Float16* w1t   = (_Float16*)(w + 2129984);
  _Float16* w2t   = (_Float16*)(w + 3768384);

  k_zero<<<1, 32, 0, stream>>>(cnt);

  for (int e = 0; e < 5; ++e){
    int nE = kH[e]*256*64;
    k_cvt<<<(nE+255)/256, 256, 0, stream>>>(emb_src[e], emb16 + embOffH[e], nE);
    int n1 = HID*kDH[e];
    k_w1t<<<(n1+255)/256, 256, 0, stream>>>(w1_src[e], w1t + w1OffH[e], kDH[e]);
    int n2 = P2H[e]*HID;
    k_w2t<<<(n2+255)/256, 256, 0, stream>>>(wr_src[e], wf_src[e], w2t + w2OffH[e], P2H[e]);
  }

  k_compact<<<NS/256, 256, 0, stream>>>(Op, fu, cnt, idx);

  IntPtrs ip;
  for (int i = 0; i < 9; ++i) ip.p[i] = (const int*)d_in[i];
  BiasPtrs bpv;
  for (int e = 0; e < 5; ++e) bpv.p[e] = b1_src[e];

  const int smem = 118528;
  hipFuncSetAttribute((const void*)k_moe,
                      hipFuncAttributeMaxDynamicSharedMemorySize, smem);
  dim3 grid(NS/64, 5);
  k_moe<<<grid, 256, smem, stream>>>(ip, bpv, cnt, idx, emb16, w1t, w2t,
                                     (float*)d_out);
}
